// Network_30262339568374
// MI455X (gfx1250) — compile-verified, matrix operand order checked by reading, not run on hardware
//
#include <hip/hip_runtime.h>
#include <hip/hip_bf16.h>
#include <math.h>

#define NN 20000
#define EE 400000
#define DD 128
#define ETILE 64   // edges per block in msg_gemm (4 WMMA subtiles)

typedef __attribute__((ext_vector_type(16))) _Float16 v16h;
typedef __attribute__((ext_vector_type(8)))  float    v8f;

// K index inside a 32-wide k-tile for element e (0..15) of a v16h fragment,
// per CDNA5 ISA 7.12.2 "16-bit A-Matrix 16x32" layout (mirrored for B).
__device__ __forceinline__ int frag_k(int e, int lane) {
    int v = e >> 1, hh = e & 1;
    int g = (lane >> 4) & 1;
    return ((v & 3) << 1) + hh + ((v & 4) << 2) + (g << 3);
}

// Repack a row-major K x Dout f32 weight matrix into per-wave WMMA B fragments.
__global__ void repack_w(const float* __restrict__ W, _Float16* __restrict__ frag,
                         int K, int Dout) {
    int ntiles = Dout >> 4;
    int f = blockIdx.x;
    int ktile = f / ntiles, ntile = f % ntiles;
    int lane = threadIdx.x;
    int n = ntile * 16 + (lane & 15);
    _Float16* dst = frag + (size_t)(f * 32 + lane) * 16;
#pragma unroll
    for (int e = 0; e < 16; ++e) {
        int k = ktile * 32 + frag_k(e, lane);
        dst[e] = (_Float16)W[k * Dout + n];
    }
}

__global__ void init_state(const float* __restrict__ s, _Float16* __restrict__ xf,
                           float* __restrict__ sfeat) {
    int idx = blockIdx.x * blockDim.x + threadIdx.x;
    if (idx >= NN * DD) return;
    int n = idx >> 7, d = idx & 127;
    xf[(size_t)n * 256 + 128 + d] = (_Float16)0.0f;   // h = 0
    if (d == 0) sfeat[n] = s[n];
}

__global__ void encode(const float* __restrict__ pos, const float* __restrict__ sfeat,
                       const float* __restrict__ Wenc, const float* __restrict__ benc,
                       _Float16* __restrict__ xf) {
    int idx = blockIdx.x * blockDim.x + threadIdx.x;
    if (idx >= NN * DD) return;
    int n = idx >> 7, d = idx & 127;
    float z = fmaf(pos[n], Wenc[d], fmaf(sfeat[n], Wenc[DD + d], benc[d]));
    xf[(size_t)n * 256 + d] = (_Float16)fmaxf(z, 0.0f);
}

__device__ __forceinline__ void flush_msg(v8f acc, int ebase, int g, int d, float bm,
                                          const int* __restrict__ dst,
                                          unsigned int* __restrict__ agg) {
#pragma unroll
    for (int r = 0; r < 8; ++r) {
        float v = fmaxf(acc[r] + bm, 0.0f);
        // agg is zero-initialized and values are relu'd (>= 0): a zero atomic is a
        // no-op, and for non-negative floats u32-max == float-max on bit patterns.
        if (v > 0.0f) {
            int ei = ebase + r + g * 8;
            atomicMax(agg + (size_t)dst[ei] * DD + d, __float_as_uint(v));
        }
    }
}

// Edge message GEMM fused with relu + scatter-max aggregation.
// One block = 64 edges (4 subtiles); 8 waves each own a 16-wide output slice.
// Loop order reuses each W_msg B fragment across 4 WMMAs.
__global__ void __launch_bounds__(256)
msg_gemm(const _Float16* __restrict__ xf, const int* __restrict__ src,
         const int* __restrict__ dst, const _Float16* __restrict__ wfrag,
         const float* __restrict__ bmsg, unsigned int* __restrict__ agg) {
    __shared__ __align__(16) _Float16 sA[ETILE * 256];   // 32 KB
    int tile = blockIdx.x;
    int t = threadIdx.x;
    // Stage 64 gathered x[src] rows: 1024 chunks of 32 B, 4 per thread.
#pragma unroll
    for (int c = 0; c < 4; ++c) {
        int chunk = t + 256 * c;
        int row = chunk >> 4, seg = chunk & 15;
        int sn = src[tile * ETILE + row];
        const uint4* gs = (const uint4*)(xf + (size_t)sn * 256 + seg * 16);
        uint4* ls = (uint4*)(sA + row * 256 + seg * 16);
        ls[0] = gs[0];
        ls[1] = gs[1];
    }
    __syncthreads();

    int lane = t & 31;
    int ntile = t >> 5;
    v8f acc0 = {}, acc1 = {}, acc2 = {}, acc3 = {};
    const _Float16* base = sA + (lane & 15) * 256;
#pragma unroll
    for (int kt = 0; kt < 8; ++kt) {
        v16h b = *(const v16h*)(wfrag + (size_t)((kt * 8 + ntile) * 32 + lane) * 16);
        v16h a0 = {}, a1 = {}, a2 = {}, a3 = {};
#pragma unroll
        for (int el = 0; el < 16; ++el) {
            int k = kt * 32 + frag_k(el, lane);
            a0[el] = base[k];
            a1[el] = base[16 * 256 + k];
            a2[el] = base[32 * 256 + k];
            a3[el] = base[48 * 256 + k];
        }
        acc0 = __builtin_amdgcn_wmma_f32_16x16x32_f16(false, a0, false, b, (short)0, acc0, false, false);
        acc1 = __builtin_amdgcn_wmma_f32_16x16x32_f16(false, a1, false, b, (short)0, acc1, false, false);
        acc2 = __builtin_amdgcn_wmma_f32_16x16x32_f16(false, a2, false, b, (short)0, acc2, false, false);
        acc3 = __builtin_amdgcn_wmma_f32_16x16x32_f16(false, a3, false, b, (short)0, acc3, false, false);
    }
    int g = (lane >> 4) & 1;
    int d = ntile * 16 + (lane & 15);
    float bm = bmsg[d];
    int eb = tile * ETILE;
    flush_msg(acc0, eb +  0, g, d, bm, dst, agg);
    flush_msg(acc1, eb + 16, g, d, bm, dst, agg);
    flush_msg(acc2, eb + 32, g, d, bm, dst, agg);
    flush_msg(acc3, eb + 48, g, d, bm, dst, agg);
}

// Node update: h = relu(x @ W_self + agg @ W_agg + b_upd); writes h-half of x in place.
__global__ void __launch_bounds__(256)
node_gemm(_Float16* __restrict__ xf, const float* __restrict__ agg,
          const _Float16* __restrict__ wself, const _Float16* __restrict__ wagg,
          const float* __restrict__ bupd) {
    __shared__ __align__(16) _Float16 sX[16 * 256];
    __shared__ __align__(16) _Float16 sG[16 * 128];
    int tile = blockIdx.x;
    int t = threadIdx.x;
    int row = t >> 4, seg = t & 15;
    int node = tile * 16 + row;
    const uint4* gx = (const uint4*)(xf + (size_t)node * 256 + seg * 16);
    uint4* lx = (uint4*)(sX + row * 256 + seg * 16);
    lx[0] = gx[0];
    lx[1] = gx[1];
    const float* ga = agg + (size_t)node * DD + seg * 8;
    _Float16* la = sG + row * 128 + seg * 8;
#pragma unroll
    for (int i = 0; i < 8; ++i) la[i] = (_Float16)ga[i];
    __syncthreads();

    int lane = t & 31;
    int ntile = t >> 5;
    v8f acc = {};
    const _Float16* xr = sX + (lane & 15) * 256;
    const _Float16* gr = sG + (lane & 15) * 128;
#pragma unroll
    for (int kt = 0; kt < 8; ++kt) {
        v16h b = *(const v16h*)(wself + (size_t)((kt * 8 + ntile) * 32 + lane) * 16);
        v16h a = {};
#pragma unroll
        for (int el = 0; el < 16; ++el)
            a[el] = xr[kt * 32 + frag_k(el, lane)];
        acc = __builtin_amdgcn_wmma_f32_16x16x32_f16(false, a, false, b,
                                                     (short)0, acc, false, false);
    }
#pragma unroll
    for (int kt = 0; kt < 4; ++kt) {
        v16h b = *(const v16h*)(wagg + (size_t)((kt * 8 + ntile) * 32 + lane) * 16);
        v16h a = {};
#pragma unroll
        for (int el = 0; el < 16; ++el)
            a[el] = gr[kt * 32 + frag_k(el, lane)];
        acc = __builtin_amdgcn_wmma_f32_16x16x32_f16(false, a, false, b,
                                                     (short)0, acc, false, false);
    }
    int g = (lane >> 4) & 1;
    int d = ntile * 16 + (lane & 15);
    float bu = bupd[d];
#pragma unroll
    for (int r = 0; r < 8; ++r) {
        int n2 = tile * 16 + r + g * 8;
        float v = fmaxf(acc[r] + bu, 0.0f);
        xf[(size_t)n2 * 256 + 128 + d] = (_Float16)v;
    }
}

// Per-edge decoder: alpha = sigmoid([h_src, h_dst] . W_dec + b); BCE + incident max.
__global__ void __launch_bounds__(256)
decode(const _Float16* __restrict__ xf, const int* __restrict__ src,
       const int* __restrict__ dst, const float* __restrict__ wdec,
       const float* __restrict__ bdec, const int* __restrict__ tgt,
       unsigned int* __restrict__ amax, float* __restrict__ lsum) {
    __shared__ float sw[256];
    sw[threadIdx.x] = wdec[threadIdx.x];
    __syncthreads();
    int e = blockIdx.x * blockDim.x + threadIdx.x;
    float bce = 0.0f;
    if (e < EE) {
        int sn = src[e], dn = dst[e];
        const v16h* hs = (const v16h*)(xf + (size_t)sn * 256 + 128);
        const v16h* hd = (const v16h*)(xf + (size_t)dn * 256 + 128);
        float acc = bdec[0];
#pragma unroll
        for (int c = 0; c < 8; ++c) {
            v16h a = hs[c], b = hd[c];
#pragma unroll
            for (int j = 0; j < 16; ++j)
                acc += (float)a[j] * sw[c * 16 + j] + (float)b[j] * sw[DD + c * 16 + j];
        }
        float alpha = 1.0f / (1.0f + __expf(-acc));
        unsigned ab = __float_as_uint(alpha);     // alpha > 0: u32 max trick valid
        atomicMax(amax + sn, ab);
        atomicMax(amax + dn, ab);
        float tv = (float)tgt[e];
        float lp = fmaxf(__logf(alpha), -100.0f);
        float l1 = fmaxf(__logf(1.0f - alpha), -100.0f);
        bce = -(tv * lp + (1.0f - tv) * l1);
    }
    for (int o = 16; o > 0; o >>= 1) bce += __shfl_down(bce, o, 32);
    if ((threadIdx.x & 31) == 0) atomicAdd(lsum, bce);
}

__global__ void ynode(const unsigned int* __restrict__ amax, const int* __restrict__ tgt,
                      float* __restrict__ sfeat, float* __restrict__ yout,
                      float* __restrict__ lsum) {
    int n = blockIdx.x * blockDim.x + threadIdx.x;
    float bce = 0.0f;
    if (n < NN) {
        float a = __uint_as_float(amax[n]);
        float y = (a >= 0.8f) ? 1.0f : 0.0f;
        sfeat[n] = y;
        if (yout) yout[n] = y;
        float tv = (float)tgt[n];
        float lp = fmaxf(__logf(y), -100.0f);          // log(0) -> -inf -> clamp
        float l1 = fmaxf(__logf(1.0f - y), -100.0f);
        bce = -(tv * lp + (1.0f - tv) * l1);
    }
    for (int o = 16; o > 0; o >>= 1) bce += __shfl_down(bce, o, 32);
    if ((threadIdx.x & 31) == 0) atomicAdd(lsum, bce);
}

__global__ void finalize(const float* __restrict__ slots, float* __restrict__ out) {
    if (threadIdx.x == 0) {
        out[NN + 0] = slots[3] / (float)EE;                              // loss_x
        out[NN + 1] = (slots[0] + slots[1] + slots[2]) / (float)EE;      // loss_h
        out[NN + 2] = slots[7] / (float)NN;                              // y_loss_x
        out[NN + 3] = (slots[4] + slots[5] + slots[6]) / (float)NN;      // y_loss_h
    }
}

extern "C" void kernel_launch(void* const* d_in, const int* in_sizes, int n_in,
                              void* d_out, int out_size, void* d_ws, size_t ws_size,
                              hipStream_t stream) {
    (void)in_sizes; (void)n_in; (void)out_size; (void)ws_size;
    const float* pos  = (const float*)d_in[0];
    const float* s    = (const float*)d_in[1];
    const int*   eidx = (const int*)d_in[2];
    const int*   pi   = (const int*)d_in[3];
    const int*   pi_h = (const int*)d_in[4];
    const int*   rch  = (const int*)d_in[5];
    const float* Wenc = (const float*)d_in[6];
    const float* benc = (const float*)d_in[7];
    const float* Wmsg = (const float*)d_in[8];
    const float* bmsg = (const float*)d_in[9];
    const float* Wslf = (const float*)d_in[10];
    const float* Wagg = (const float*)d_in[11];
    const float* bupd = (const float*)d_in[12];
    const float* Wdec = (const float*)d_in[13];
    const float* bdec = (const float*)d_in[14];
    const int* src  = eidx;
    const int* dstp = eidx + EE;
    float* out = (float*)d_out;

    char* w = (char*)d_ws;
    _Float16* xf    = (_Float16*)w;  w += (size_t)NN * 256 * sizeof(_Float16);  // [N,256] f16: [z | h]
    float*    agg   = (float*)w;     w += (size_t)NN * DD * sizeof(float);      // scatter-max accum
    unsigned* amax  = (unsigned*)w;  w += (size_t)NN * sizeof(unsigned);        // per-node alpha max
    float*    sfeat = (float*)w;     w += (size_t)NN * sizeof(float);
    _Float16* fmsg  = (_Float16*)w;  w += (size_t)256 * DD * sizeof(_Float16);  // WMMA B fragments
    _Float16* fslf  = (_Float16*)w;  w += (size_t)256 * DD * sizeof(_Float16);
    _Float16* fagg  = (_Float16*)w;  w += (size_t)DD * DD * sizeof(_Float16);
    float*    slots = (float*)w;     // 8 loss accumulators

    repack_w<<<(256 / 32) * (DD / 16), 32, 0, stream>>>(Wmsg, fmsg, 256, DD);
    repack_w<<<(256 / 32) * (DD / 16), 32, 0, stream>>>(Wslf, fslf, 256, DD);
    repack_w<<<(DD / 32) * (DD / 16), 32, 0, stream>>>(Wagg, fagg, DD, DD);
    hipMemsetAsync(slots, 0, 8 * sizeof(float), stream);
    init_state<<<(NN * DD + 255) / 256, 256, 0, stream>>>(s, xf, sfeat);

    for (int iter = 0; iter < 4; ++iter) {
        hipMemsetAsync(agg, 0, (size_t)NN * DD * sizeof(float), stream);
        hipMemsetAsync(amax, 0, (size_t)NN * sizeof(unsigned), stream);
        encode<<<(NN * DD + 255) / 256, 256, 0, stream>>>(pos, sfeat, Wenc, benc, xf);
        msg_gemm<<<EE / ETILE, 256, 0, stream>>>(xf, src, dstp, fmsg, bmsg, (unsigned*)agg);
        node_gemm<<<NN / 16, 256, 0, stream>>>(xf, agg, fslf, fagg, bupd);
        const int* etgt = (iter < 3) ? (pi_h + (size_t)(iter + 1) * EE) : pi;
        decode<<<(EE + 255) / 256, 256, 0, stream>>>(xf, src, dstp, Wdec, bdec, etgt,
                                                     amax, slots + iter);
        const int* ntgt = rch + (size_t)((iter < 3) ? iter + 1 : 3) * NN;
        ynode<<<(NN + 255) / 256, 256, 0, stream>>>(amax, ntgt, sfeat,
                                                    (iter == 3) ? out : nullptr,
                                                    slots + 4 + iter);
    }
    finalize<<<1, 1, 0, stream>>>(slots, out);
}
